// ContextualLoss_79233556676703
// MI455X (gfx1250) — matrix operand order repaired and checked by reading, not compiled, not run on hardware
//
#include <hip/hip_runtime.h>
#include <hip/hip_bf16.h>

typedef __attribute__((ext_vector_type(16))) _Float16 v16h;
typedef __attribute__((ext_vector_type(8)))  _Float16 v8h;
typedef __attribute__((ext_vector_type(8)))  float    v8f;

union V16 { v16h v; v8h h[2]; };

#define BATCH 8
#define BM 128
#define BN 128
#define KC 64
#define PAD 8
#define LROW (KC + PAD)
#define HW1 4096
#define HW2 1024
#define C1 128
#define C2 256

// ---------------------------------------------------------------------------
// Kernel 1: column-normalize (over C) and transpose to [b][pos][C] in f16.
// ---------------------------------------------------------------------------
__global__ __launch_bounds__(256)
void cx_normalize_kernel(const float* __restrict__ X, _Float16* __restrict__ Xt,
                         int C, int HW) {
    const int b = blockIdx.y;
    const int j = blockIdx.x * blockDim.x + threadIdx.x;   // HW multiple of 256
    const float* Xb = X + (size_t)b * C * HW;
    float s = 0.0f;
    for (int c = 0; c < C; ++c) {
        float v = Xb[(size_t)c * HW + j];
        s += v * v;
    }
    const float inv = 1.0f / (sqrtf(s) + 1e-5f);
    _Float16* out = Xt + ((size_t)b * HW + j) * C;
    for (int c0 = 0; c0 < C; c0 += 8) {
        v8h p;
#pragma unroll
        for (int t = 0; t < 8; ++t)
            p[t] = (_Float16)(Xb[(size_t)(c0 + t) * HW + j] * inv);
        *(v8h*)&out[c0] = p;
    }
}

// ---------------------------------------------------------------------------
// Async global->LDS staging: each thread DMAs 16B per issue (ASYNCcnt path).
// Counted/unrolled so the trip count is uniform -> no EXEC-mask loop dance.
// ---------------------------------------------------------------------------
template <int ROWS>
__device__ __forceinline__ void stage_async(const _Float16* __restrict__ g,
                                            unsigned ldsBase, int k0,
                                            int cdim, int tid) {
    constexpr int ITERS = (ROWS * KC) / (256 * 8);   // 4 for ROWS=128
#pragma unroll
    for (int it = 0; it < ITERS; ++it) {
        const int base = tid * 8 + it * 256 * 8;
        const int r = base / KC, c = base % KC;
        const unsigned loff = ldsBase + (unsigned)((r * LROW + c) * 2);
        const unsigned long long ga =
            (unsigned long long)(uintptr_t)(g + (size_t)r * cdim + k0 + c);
        asm volatile("global_load_async_to_lds_b128 %0, %1, off"
                     :: "v"(loff), "v"(ga) : "memory");
    }
}
__device__ __forceinline__ void wait_async0() {
    asm volatile("s_wait_asynccnt 0x0" ::: "memory");
}

// ---------------------------------------------------------------------------
// Kernel 2/3: fused WMMA GEMM + row reduction, async double-buffered K chunks.
//   pass 0: dmin[b][i] = min_j (1 - sim[i][j])
//   pass 1: cx[b][i]   = w_max / (sum_j exp((1 - d*inv)/h) + eps)  (over dmin buf)
// 256 threads = 8 waves as 4(M) x 2(N); wave tile = 32 rows x 64 cols
// (2 M-tiles x 4 N-tiles = 8 WMMA accumulators per wave).
// ---------------------------------------------------------------------------
template <int CDIM>
__global__ __launch_bounds__(256)
void cx_gemm_kernel(const _Float16* __restrict__ At,   // (B, HW, CDIM) f16
                    const _Float16* __restrict__ Bt,   // (B, HW, CDIM) f16
                    float* __restrict__ dmin,          // (B, HW)
                    int HW, int pass) {
    __shared__ _Float16 Al[2][BM][LROW];
    __shared__ _Float16 Bl[2][BN][LROW];
    __shared__ float    part[2][BM];

    const int tid  = threadIdx.x;
    const int lane = tid & 31;
    const int wave = tid >> 5;
    const int wm   = wave >> 1;     // 0..3 (M)
    const int wn   = wave & 1;      // 0..1 (N)
    const int hl   = lane >> 4;     // half-wave
    const int l16  = lane & 15;
    const int b    = blockIdx.y;
    const int i0   = blockIdx.x * BM;

    const float INVH = 1.0f / 0.3f;
    const float EPSF = 1e-5f;

    const _Float16* Ag = At + ((size_t)b * HW + i0) * CDIM;

    const unsigned ldsA0 = (unsigned)(uintptr_t)&Al[0][0][0];
    const unsigned ldsA1 = (unsigned)(uintptr_t)&Al[1][0][0];
    const unsigned ldsB0 = (unsigned)(uintptr_t)&Bl[0][0][0];
    const unsigned ldsB1 = (unsigned)(uintptr_t)&Bl[1][0][0];

    // C-tile layout: lane L, vgpr v -> row = v + 8*(L/16), col = L%16.
    float racc[2][8];
    float invd[2][8];
#pragma unroll
    for (int mt = 0; mt < 2; ++mt)
#pragma unroll
        for (int v = 0; v < 8; ++v) racc[mt][v] = (pass == 0) ? 3.4e38f : 0.0f;
    if (pass != 0) {
#pragma unroll
        for (int mt = 0; mt < 2; ++mt)
#pragma unroll
            for (int v = 0; v < 8; ++v) {
                float dm = dmin[(size_t)b * HW + i0 + wm * 32 + mt * 16 + v + 8 * hl];
                invd[mt][v] = 1.0f / (dm + EPSF);
            }
    }

    for (int j0 = 0; j0 < HW; j0 += BN) {
        const _Float16* Bg = Bt + ((size_t)b * HW + j0) * CDIM;
        v8f acc[2][4] = {};

        // prologue: chunk 0 -> buffer 0
        stage_async<BM>(Ag, ldsA0, 0, CDIM, tid);
        stage_async<BN>(Bg, ldsB0, 0, CDIM, tid);
        wait_async0();
        __syncthreads();

        int buf = 0;
        for (int k0 = 0; k0 < CDIM; k0 += KC) {
            if (k0 + KC < CDIM) {    // uniform branch: prefetch next chunk
                stage_async<BM>(Ag, buf ? ldsA0 : ldsA1, k0 + KC, CDIM, tid);
                stage_async<BN>(Bg, buf ? ldsB0 : ldsB1, k0 + KC, CDIM, tid);
            }
#pragma unroll
            for (int kk = 0; kk < KC; kk += 32) {
                V16 a[2];
#pragma unroll
                for (int mt = 0; mt < 2; ++mt) {
                    const int mr = wm * 32 + mt * 16 + l16;
                    a[mt].h[0] = *(const v8h*)&Al[buf][mr][kk + 8 * hl];
                    a[mt].h[1] = *(const v8h*)&Al[buf][mr][kk + 16 + 8 * hl];
                }
#pragma unroll
                for (int nt = 0; nt < 4; ++nt) {
                    const int nr = wn * 64 + nt * 16 + l16;
                    V16 bf;
                    bf.h[0] = *(const v8h*)&Bl[buf][nr][kk + 8 * hl];
                    bf.h[1] = *(const v8h*)&Bl[buf][nr][kk + 16 + 8 * hl];
#pragma unroll
                    for (int mt = 0; mt < 2; ++mt)
                        acc[mt][nt] = __builtin_amdgcn_wmma_f32_16x16x32_f16(
                            false, a[mt].v, false, bf.v, (short)0, acc[mt][nt],
                            false, false);
                }
            }
            wait_async0();           // my prefetch stores landed in LDS
            __syncthreads();         // everyone done reading buf & staging buf^1
            buf ^= 1;
        }

        // fold this N-block into per-row accumulators
        if (pass == 0) {
#pragma unroll
            for (int mt = 0; mt < 2; ++mt)
#pragma unroll
                for (int nt = 0; nt < 4; ++nt)
#pragma unroll
                    for (int v = 0; v < 8; ++v)
                        racc[mt][v] = fminf(racc[mt][v], 1.0f - acc[mt][nt][v]);
        } else {
#pragma unroll
            for (int mt = 0; mt < 2; ++mt)
#pragma unroll
                for (int nt = 0; nt < 4; ++nt)
#pragma unroll
                    for (int v = 0; v < 8; ++v) {
                        float d = 1.0f - acc[mt][nt][v];
                        racc[mt][v] += __expf((1.0f - d * invd[mt][v]) * INVH);
                    }
        }
    }

    // reduce across 16 column-lanes (stays within each half-wave)
#pragma unroll
    for (int mt = 0; mt < 2; ++mt)
#pragma unroll
        for (int v = 0; v < 8; ++v) {
            float x = racc[mt][v];
#pragma unroll
            for (int m = 1; m <= 8; m <<= 1) {
                float o = __shfl_xor(x, m, 32);
                x = (pass == 0) ? fminf(x, o) : (x + o);
            }
            racc[mt][v] = x;
        }
    if (l16 == 0) {
#pragma unroll
        for (int mt = 0; mt < 2; ++mt)
#pragma unroll
            for (int v = 0; v < 8; ++v)
                part[wn][wm * 32 + mt * 16 + v + 8 * hl] = racc[mt][v];
    }
    __syncthreads();

    // combine the two wave_n halves and write the per-row result
    if (tid < BM) {
        const size_t gi = (size_t)b * HW + i0 + tid;
        if (pass == 0) {
            dmin[gi] = fminf(part[0][tid], part[1][tid]);
        } else {
            float dm   = dmin[gi];
            float inv  = 1.0f / (dm + EPSF);
            float wmax = __expf((1.0f - dm * inv) * INVH);
            dmin[gi]   = wmax / (part[0][tid] + part[1][tid] + EPSF);  // cx_row
        }
    }
}

// ---------------------------------------------------------------------------
// Kernel 4: deterministic per-sample sum of cx rows.
// ---------------------------------------------------------------------------
__global__ __launch_bounds__(256)
void cx_reduce_kernel(const float* __restrict__ cx, int HW, float* __restrict__ sums) {
    __shared__ float sm[256];
    const int b = blockIdx.x;
    float s = 0.0f;
    for (int i = threadIdx.x; i < HW; i += 256) s += cx[(size_t)b * HW + i];
    sm[threadIdx.x] = s;
    __syncthreads();
    for (int off = 128; off > 0; off >>= 1) {
        if (threadIdx.x < off) sm[threadIdx.x] += sm[threadIdx.x + off];
        __syncthreads();
    }
    if (threadIdx.x == 0) sums[b] = sm[0];
}

// ---------------------------------------------------------------------------
// Kernel 5: scalar finalize.
// ---------------------------------------------------------------------------
__global__ void cx_finalize_kernel(const float* __restrict__ sums, float* __restrict__ out) {
    float l1 = 0.0f, l2 = 0.0f;
    for (int s = 0; s < 8; ++s) l1 += -logf(sums[s] * (1.0f / (float)HW1) + 1e-5f);
    for (int s = 0; s < 8; ++s) l2 += -logf(sums[8 + s] * (1.0f / (float)HW2) + 1e-5f);
    out[0] = 0.5f * (l1 / 8.0f + l2 / 8.0f);
}

// ---------------------------------------------------------------------------
extern "C" void kernel_launch(void* const* d_in, const int* in_sizes, int n_in,
                              void* d_out, int out_size, void* d_ws, size_t ws_size,
                              hipStream_t stream) {
    const float* fx1 = (const float*)d_in[0];
    const float* fy1 = (const float*)d_in[1];
    const float* fx2 = (const float*)d_in[2];
    const float* fy2 = (const float*)d_in[3];

    const size_t N1 = (size_t)BATCH * HW1 * C1;   // 4,194,304 halfs
    const size_t N2 = (size_t)BATCH * HW2 * C2;   // 2,097,152 halfs

    _Float16* A1t = (_Float16*)d_ws;
    _Float16* B1t = A1t + N1;
    _Float16* A2t = B1t + N1;
    _Float16* B2t = A2t + N2;
    float* dmin1 = (float*)(B2t + N2);            // (B, HW1); pass1 rewrites with cx
    float* dmin2 = dmin1 + (size_t)BATCH * HW1;   // (B, HW2)
    float* sums  = dmin2 + (size_t)BATCH * HW2;   // 16 floats

    // normalize + transpose to f16 [b][pos][C]
    cx_normalize_kernel<<<dim3(HW1 / 256, BATCH), 256, 0, stream>>>(fx1, A1t, C1, HW1);
    cx_normalize_kernel<<<dim3(HW1 / 256, BATCH), 256, 0, stream>>>(fy1, B1t, C1, HW1);
    cx_normalize_kernel<<<dim3(HW2 / 256, BATCH), 256, 0, stream>>>(fx2, A2t, C2, HW2);
    cx_normalize_kernel<<<dim3(HW2 / 256, BATCH), 256, 0, stream>>>(fy2, B2t, C2, HW2);

    // layer 1: d_min pass, then exp-sum/cx pass
    cx_gemm_kernel<C1><<<dim3(HW1 / BM, BATCH), 256, 0, stream>>>(A1t, B1t, dmin1, HW1, 0);
    cx_gemm_kernel<C1><<<dim3(HW1 / BM, BATCH), 256, 0, stream>>>(A1t, B1t, dmin1, HW1, 1);
    // layer 2
    cx_gemm_kernel<C2><<<dim3(HW2 / BM, BATCH), 256, 0, stream>>>(A2t, B2t, dmin2, HW2, 0);
    cx_gemm_kernel<C2><<<dim3(HW2 / BM, BATCH), 256, 0, stream>>>(A2t, B2t, dmin2, HW2, 1);

    // deterministic reductions + finalize
    cx_reduce_kernel<<<BATCH, 256, 0, stream>>>(dmin1, HW1, sums);
    cx_reduce_kernel<<<BATCH, 256, 0, stream>>>(dmin2, HW2, sums + 8);
    cx_finalize_kernel<<<1, 1, 0, stream>>>(sums, (float*)d_out);
}
